// DIFFCrossAttention_90031104459370
// MI455X (gfx1250) — matrix operand order repaired and checked by reading, not compiled, not run on hardware
//
#include <hip/hip_runtime.h>
#include <cstdint>

#define DIM_ 1024
#define H_ 16
#define HD_ 64
#define NQ_ 1024
#define NK_ 2048
#define B_ 2
#define SCALE_F 0.125f          // HD^-0.5 = 1/8
#define LAMBDA_INIT_F 0.1f

typedef __attribute__((ext_vector_type(16))) __bf16   bf16x16;
typedef __attribute__((ext_vector_type(8)))  float    f32x8;
typedef __attribute__((ext_vector_type(4)))  uint32_t u32x4;

union FragU {
    bf16x16 v;
    u32x4   q[2];
};

__device__ __forceinline__ uint16_t f2bf(float f) {
    uint32_t u = __builtin_bit_cast(uint32_t, f);
    u += 0x7fffu + ((u >> 16) & 1u);          // round to nearest even
    return (uint16_t)(u >> 16);
}

__device__ __forceinline__ f32x8 zero8() {
    f32x8 z;
#pragma unroll
    for (int i = 0; i < 8; ++i) z[i] = 0.0f;
    return z;
}

__device__ __forceinline__ f32x8 wmma_bf16(const FragU& a, const FragU& b, f32x8 c) {
    return __builtin_amdgcn_wmma_f32_16x16x32_bf16(
        /*neg_a=*/false, a.v, /*neg_b=*/false, b.v,
        /*c_mod=*/(short)0, c, /*reuse_a=*/false, /*reuse_b=*/false);
}

// CDNA5 async copy: global -> LDS, 16B per lane, tracked by ASYNCcnt.
// Generic pointers to __shared__ objects carry the LDS byte offset in their
// low 32 bits (aperture rules, ISA 10.2), which is what VDST wants.
__device__ __forceinline__ void async_g2l_b128(void* lds_ptr, const void* gptr) {
    uint32_t loff = (uint32_t)(uintptr_t)lds_ptr;
    uint64_t gaddr = (uint64_t)(uintptr_t)gptr;
    asm volatile("global_load_async_to_lds_b128 %0, %1, off"
                 :: "v"(loff), "v"(gaddr)
                 : "memory");
}

__device__ __forceinline__ void wait_async0() {
    asm volatile("s_wait_asynccnt 0" ::: "memory");
}

// Partial wait: ASYNCcnt completes in order, so cnt<=4 means everything older
// than the newest tile (4 async ops per wave) has landed in LDS.
__device__ __forceinline__ void wait_async4() {
    asm volatile("s_wait_asynccnt 4" ::: "memory");
}

// ---------------------------------------------------------------------------
// fp32 -> bf16 conversion (packs 2 per dword)
// ---------------------------------------------------------------------------
__global__ void cvt_kernel(const float* __restrict__ in,
                           uint32_t* __restrict__ out, int n2) {
    int i = blockIdx.x * blockDim.x + threadIdx.x;
    if (i < n2) {
        float a = in[2 * i];
        float b = in[2 * i + 1];
        out[i] = (uint32_t)f2bf(a) | ((uint32_t)f2bf(b) << 16);
    }
}

// ---------------------------------------------------------------------------
// Generic bf16 GEMM: C[M,N] = A[M,K=1024] * W[N,K=1024]^T
// 128x128 tile, 8 waves (4x2), each wave 32x64 via 2x4 16x16x32 WMMA accums.
// Triple-buffered LDS filled by async global->LDS copies: two tiles kept in
// flight, partial s_wait_asynccnt 4 retires only the oldest one.
//   mode 0: q-style  -> dst0 bf16 [B,H,seq,HD]
//   mode 1: kv1      -> k half dst0 bf16 [B,H,seq,HD]; v half dst1 bf16 [B,H,HD,seq]
//   mode 2: k2       -> dst0 bf16 [B,H,seq,HD]
//   mode 3: proj     -> outf f32 row-major [M,N] + bias
// ---------------------------------------------------------------------------
__global__ __launch_bounds__(256) void gemm_kernel(
    const uint16_t* __restrict__ A, const uint16_t* __restrict__ W,
    int M, int N, int mode, int seq,
    uint16_t* __restrict__ dst0, uint16_t* __restrict__ dst1,
    float* __restrict__ outf, const float* __restrict__ bias)
{
    const int K = 1024;
    const int LS = 40;                       // padded LDS row stride (bank-conflict free)
    __shared__ uint16_t As[3][128 * 40];
    __shared__ uint16_t Ws[3][128 * 40];

    const int tid  = threadIdx.x;
    const int wave = tid >> 5, lane = tid & 31;
    const int half = lane >> 4, ln = lane & 15;
    const int wm = wave >> 1, wn = wave & 1;
    const int rb = blockIdx.y * 128, cb = blockIdx.x * 128;

    f32x8 acc[2][4];
#pragma unroll
    for (int mi = 0; mi < 2; ++mi)
#pragma unroll
        for (int nj = 0; nj < 4; ++nj) acc[mi][nj] = zero8();

    auto load_tile = [&](int kb, int buf) {   // 4 async ops per wave
#pragma unroll
        for (int it = 0; it < 2; ++it) {
            int id = tid + 256 * it;
            int row = id >> 2, ch = id & 3;
            async_g2l_b128(&As[buf][row * LS + ch * 8],
                           &A[(size_t)(rb + row) * K + kb + ch * 8]);
            async_g2l_b128(&Ws[buf][row * LS + ch * 8],
                           &W[(size_t)(cb + row) * K + kb + ch * 8]);
        }
    };

    load_tile(0, 0);
    load_tile(32, 1);
    wait_async4();              // tile 0 landed; tile 1 still in flight
    __syncthreads();

    for (int kb = 0; kb < K; kb += 32) {
        const int step = kb >> 5;
        const int cur  = step % 3;
        const bool pre = (kb + 64 < K);
        if (pre) load_tile(kb + 64, (cur + 2) % 3);   // keep two tiles in flight

        FragU af[2], bfr[4];
#pragma unroll
        for (int mi = 0; mi < 2; ++mi) {
            int row = wm * 32 + mi * 16 + ln;
            int k0  = half ? 8 : 0;
            af[mi].q[0] = *(const u32x4*)&As[cur][row * LS + k0];
            af[mi].q[1] = *(const u32x4*)&As[cur][row * LS + k0 + 16];
        }
#pragma unroll
        for (int nj = 0; nj < 4; ++nj) {
            int row = wn * 64 + nj * 16 + ln;
            int k0  = half ? 8 : 0;
            bfr[nj].q[0] = *(const u32x4*)&Ws[cur][row * LS + k0];
            bfr[nj].q[1] = *(const u32x4*)&Ws[cur][row * LS + k0 + 16];
        }
#pragma unroll
        for (int mi = 0; mi < 2; ++mi)
#pragma unroll
            for (int nj = 0; nj < 4; ++nj)
                acc[mi][nj] = wmma_bf16(af[mi], bfr[nj], acc[mi][nj]);

        // Retire the oldest in-flight tile (the one read next iteration);
        // when nothing new was issued, drain fully.
        if (pre) wait_async4(); else wait_async0();
        __syncthreads();
    }

#pragma unroll
    for (int mi = 0; mi < 2; ++mi) {
#pragma unroll
        for (int nj = 0; nj < 4; ++nj) {
#pragma unroll
            for (int r = 0; r < 8; ++r) {
                float v = acc[mi][nj][r];
                int gr = rb + wm * 32 + mi * 16 + r + half * 8;
                int gc = cb + wn * 64 + nj * 16 + ln;
                if (mode == 3) {
                    outf[(size_t)gr * N + gc] = v + bias[gc];
                } else if (mode == 0 || mode == 2) {
                    int b = gr / seq, s = gr % seq;
                    int h = gc >> 6, d = gc & 63;
                    dst0[((size_t)(b * H_ + h) * seq + s) * HD_ + d] = f2bf(v);
                } else {  // mode 1: kv1 -> k1 + transposed v1
                    int b = gr / seq, s = gr % seq;
                    int p = gc >> 10;
                    int h = (gc >> 6) & 15, d = gc & 63;
                    if (p == 0)
                        dst0[((size_t)(b * H_ + h) * seq + s) * HD_ + d] = f2bf(v);
                    else
                        dst1[((size_t)(b * H_ + h) * HD_ + d) * seq + s] = f2bf(v);
                }
            }
        }
    }
}

// ---------------------------------------------------------------------------
// Differential dual flash attention.
// Grid: B*H*(NQ/128) workgroups; 256 threads = 8 waves; wave owns 16 q rows.
// Streams NK in 64-key blocks: K1/K2/V^T tiles arrive via async global->LDS;
// S via WMMA; online softmax with half-wave butterflies; P bounces through a
// wave-private LDS slab (no barrier needed: per-wave LDS ops are in-order);
// O accumulated via WMMA against the shared transposed V tile.
// ---------------------------------------------------------------------------
__global__ __launch_bounds__(256) void diffattn_kernel(
    const uint16_t* __restrict__ q1b, const uint16_t* __restrict__ q2b,
    const uint16_t* __restrict__ k1b, const uint16_t* __restrict__ k2b,
    const uint16_t* __restrict__ v1t,
    const float* __restrict__ lam1, const float* __restrict__ lam2,
    float* __restrict__ xbuf)
{
    const int LS = 72;                       // padded row stride, conflict-free
    __shared__ uint16_t k1s[64 * 72];
    __shared__ uint16_t k2s[64 * 72];
    __shared__ uint16_t vts[64 * 72];        // [d][key]
    __shared__ uint16_t ps[8][16 * 72];      // per-wave P slab (16 x 64 bf16)

    const int tid  = threadIdx.x;
    const int wave = tid >> 5, lane = tid & 31;
    const int half = lane >> 4, ln = lane & 15;
    const int tiles = NQ_ / 128;
    const int bh = blockIdx.x / tiles;
    const int qt = blockIdx.x % tiles;
    const int h  = bh & 15;
    const int b  = bh >> 4;

    const uint16_t* q1p = q1b + ((size_t)bh * NQ_ + qt * 128) * HD_;
    const uint16_t* q2p = q2b + ((size_t)bh * NQ_ + qt * 128) * HD_;
    const uint16_t* k1p = k1b + (size_t)bh * NK_ * HD_;
    const uint16_t* k2p = k2b + (size_t)bh * NK_ * HD_;
    const uint16_t* vtp = v1t + (size_t)bh * HD_ * NK_;

    // Preload per-wave Q fragments (16 rows x 64 HD = 2 A-frags each)
    FragU qa1[2], qa2[2];
    {
        int row = wave * 16 + ln;
#pragma unroll
        for (int c = 0; c < 2; ++c) {
            int k0 = c * 32 + (half ? 8 : 0);
            qa1[c].q[0] = *(const u32x4*)&q1p[row * HD_ + k0];
            qa1[c].q[1] = *(const u32x4*)&q1p[row * HD_ + k0 + 16];
            qa2[c].q[0] = *(const u32x4*)&q2p[row * HD_ + k0];
            qa2[c].q[1] = *(const u32x4*)&q2p[row * HD_ + k0 + 16];
        }
    }

    f32x8 O1[4], O2[4];
    float m1[8], l1[8], m2[8], l2[8];
#pragma unroll
    for (int j = 0; j < 4; ++j) { O1[j] = zero8(); O2[j] = zero8(); }
#pragma unroll
    for (int r = 0; r < 8; ++r) { m1[r] = m2[r] = -3.0e38f; l1[r] = l2[r] = 0.f; }

    // One softmax-attention stream: S = Q K^T (WMMA), online softmax,
    // P -> wave-private LDS -> A-frags, O += P V (WMMA).
    auto stream = [&](const uint16_t* ks, const FragU* qa,
                      float* m, float* l, f32x8* O) {
        f32x8 S[4];
#pragma unroll
        for (int j = 0; j < 4; ++j) {
            FragU b0, b1;
            int rw = j * 16 + ln;
            int k0 = half ? 8 : 0;
            b0.q[0] = *(const u32x4*)&ks[rw * LS + k0];
            b0.q[1] = *(const u32x4*)&ks[rw * LS + k0 + 16];
            b1.q[0] = *(const u32x4*)&ks[rw * LS + 32 + k0];
            b1.q[1] = *(const u32x4*)&ks[rw * LS + 32 + k0 + 16];
            f32x8 s = wmma_bf16(qa[0], b0, zero8());
            s = wmma_bf16(qa[1], b1, s);
            S[j] = s * SCALE_F;
        }
#pragma unroll
        for (int r = 0; r < 8; ++r) {
            float v = fmaxf(fmaxf(S[0][r], S[1][r]), fmaxf(S[2][r], S[3][r]));
#pragma unroll
            for (int mk = 1; mk < 16; mk <<= 1) v = fmaxf(v, __shfl_xor(v, mk, 32));
            float mnew  = fmaxf(m[r], v);
            float alpha = __expf(m[r] - mnew);
            m[r] = mnew;
            float rs = 0.f;
#pragma unroll
            for (int j = 0; j < 4; ++j) {
                float p = __expf(S[j][r] - mnew);
                S[j][r] = p;
                rs += p;
            }
#pragma unroll
            for (int mk = 1; mk < 16; mk <<= 1) rs += __shfl_xor(rs, mk, 32);
            l[r] = l[r] * alpha + rs;
#pragma unroll
            for (int j = 0; j < 4; ++j) O[j][r] *= alpha;
        }
#pragma unroll
        for (int j = 0; j < 4; ++j)
#pragma unroll
            for (int r = 0; r < 8; ++r)
                ps[wave][(r + half * 8) * LS + j * 16 + ln] = f2bf(S[j][r]);

        // Same-wave LDS ops are in-order: safe to read P back without barrier.
#pragma unroll
        for (int c = 0; c < 2; ++c) {
            FragU pa;
            int k0 = c * 32 + (half ? 8 : 0);
            pa.q[0] = *(const u32x4*)&ps[wave][ln * LS + k0];
            pa.q[1] = *(const u32x4*)&ps[wave][ln * LS + k0 + 16];
#pragma unroll
            for (int j = 0; j < 4; ++j) {
                FragU vb;
                int rw = j * 16 + ln;
                int kk = c * 32 + (half ? 8 : 0);
                vb.q[0] = *(const u32x4*)&vts[rw * LS + kk];
                vb.q[1] = *(const u32x4*)&vts[rw * LS + kk + 16];
                O[j] = wmma_bf16(pa, vb, O[j]);
            }
        }
    };

    for (int kb = 0; kb < NK_; kb += 64) {
        // async global -> LDS tile fill (64 keys of K1, K2, V^T)
#pragma unroll
        for (int it = 0; it < 2; ++it) {
            int id = tid + 256 * it;
            int row = id >> 3, ch = id & 7;
            async_g2l_b128(&k1s[row * LS + ch * 8],
                           &k1p[(size_t)(kb + row) * HD_ + ch * 8]);
            async_g2l_b128(&k2s[row * LS + ch * 8],
                           &k2p[(size_t)(kb + row) * HD_ + ch * 8]);
            async_g2l_b128(&vts[row * LS + ch * 8],
                           &vtp[(size_t)row * NK_ + kb + ch * 8]);
        }
        wait_async0();
        __syncthreads();

        stream(k1s, qa1, m1, l1, O1);
        stream(k2s, qa2, m2, l2, O2);

        __syncthreads();        // all waves done reading tiles before refill
    }

    const float lam = lam1[h] - lam2[h] + LAMBDA_INIT_F;
#pragma unroll
    for (int j = 0; j < 4; ++j) {
#pragma unroll
        for (int r = 0; r < 8; ++r) {
            float o = O1[j][r] / l1[r] - lam * (O2[j][r] / l2[r]);
            int q   = qt * 128 + wave * 16 + r + half * 8;
            int col = h * HD_ + j * 16 + ln;
            xbuf[((size_t)b * NQ_ + q) * DIM_ + col] = o;
        }
    }
}

// ---------------------------------------------------------------------------
// RMSNorm over DIM, writes bf16 normalized activations for the final GEMM.
// ---------------------------------------------------------------------------
__global__ __launch_bounds__(256) void rmsnorm_kernel(
    const float* __restrict__ x, const float* __restrict__ w,
    uint16_t* __restrict__ xn)
{
    __shared__ float red[256];
    const int row = blockIdx.x, tid = threadIdx.x;
    const float* xr = x + (size_t)row * DIM_;
    float ss = 0.f;
    for (int i = tid; i < DIM_; i += 256) { float v = xr[i]; ss += v * v; }
    red[tid] = ss;
    __syncthreads();
    for (int s = 128; s > 0; s >>= 1) {
        if (tid < s) red[tid] += red[tid + s];
        __syncthreads();
    }
    float inv = rsqrtf(red[0] * (1.0f / DIM_) + 1e-6f);
    for (int i = tid; i < DIM_; i += 256)
        xn[(size_t)row * DIM_ + i] = f2bf(xr[i] * inv * w[i]);
}

// ---------------------------------------------------------------------------
extern "C" void kernel_launch(void* const* d_in, const int* in_sizes, int n_in,
                              void* d_out, int out_size, void* d_ws, size_t ws_size,
                              hipStream_t stream) {
    (void)in_sizes; (void)n_in; (void)out_size; (void)ws_size;
    const float* query  = (const float*)d_in[0];
    const float* key    = (const float*)d_in[1];
    // d_in[2] value, d_in[3] qpos, d_in[4] kpos: unused by the reference.
    const float* q1_w   = (const float*)d_in[5];
    const float* q2_w   = (const float*)d_in[6];
    const float* kv1_w  = (const float*)d_in[7];
    const float* kv2_w  = (const float*)d_in[8];
    const float* proj_w = (const float*)d_in[9];
    const float* proj_b = (const float*)d_in[10];
    const float* norm_w = (const float*)d_in[11];
    const float* lam1   = (const float*)d_in[12];
    const float* lam2   = (const float*)d_in[13];
    float* out = (float*)d_out;

    char* ws = (char*)d_ws;
    size_t off = 0;
    auto alloc = [&](size_t bytes) {
        char* p = ws + off;
        off += (bytes + 255) & ~(size_t)255;
        return p;
    };
    uint16_t* qbf  = (uint16_t*)alloc(2048ull * 1024 * 2);
    uint16_t* kbf  = (uint16_t*)alloc(4096ull * 1024 * 2);
    uint16_t* wq1  = (uint16_t*)alloc(1024ull * 1024 * 2);
    uint16_t* wq2  = (uint16_t*)alloc(1024ull * 1024 * 2);
    uint16_t* wkv1 = (uint16_t*)alloc(2048ull * 1024 * 2);
    uint16_t* wk2  = (uint16_t*)alloc(1024ull * 1024 * 2);
    uint16_t* wpr  = (uint16_t*)alloc(1024ull * 1024 * 2);
    uint16_t* q1b  = (uint16_t*)alloc((size_t)B_ * H_ * NQ_ * HD_ * 2);
    uint16_t* q2b  = (uint16_t*)alloc((size_t)B_ * H_ * NQ_ * HD_ * 2);
    uint16_t* k1b  = (uint16_t*)alloc((size_t)B_ * H_ * NK_ * HD_ * 2);
    uint16_t* k2b  = (uint16_t*)alloc((size_t)B_ * H_ * NK_ * HD_ * 2);
    uint16_t* v1t  = (uint16_t*)alloc((size_t)B_ * H_ * NK_ * HD_ * 2);
    float*    xbuf = (float*)   alloc(2048ull * 1024 * 4);
    uint16_t* xn   = (uint16_t*)alloc(2048ull * 1024 * 2);

    auto cvt = [&](const float* src, uint16_t* dst, size_t n) {
        int n2 = (int)(n / 2);
        cvt_kernel<<<(n2 + 255) / 256, 256, 0, stream>>>(src, (uint32_t*)dst, n2);
    };
    cvt(query,  qbf,  2048ull * 1024);
    cvt(key,    kbf,  4096ull * 1024);
    cvt(q1_w,   wq1,  1024ull * 1024);
    cvt(q2_w,   wq2,  1024ull * 1024);
    cvt(kv1_w,  wkv1, 2048ull * 1024);
    cvt(kv2_w,  wk2,  1024ull * 1024);   // only the k2 half (first DIM rows) is used
    cvt(proj_w, wpr,  1024ull * 1024);

    dim3 blk(256);
    gemm_kernel<<<dim3(1024 / 128, 2048 / 128), blk, 0, stream>>>(
        qbf, wq1, 2048, 1024, 0, NQ_, q1b, nullptr, nullptr, nullptr);
    gemm_kernel<<<dim3(1024 / 128, 2048 / 128), blk, 0, stream>>>(
        qbf, wq2, 2048, 1024, 0, NQ_, q2b, nullptr, nullptr, nullptr);
    gemm_kernel<<<dim3(2048 / 128, 4096 / 128), blk, 0, stream>>>(
        kbf, wkv1, 4096, 2048, 1, NK_, k1b, v1t, nullptr, nullptr);
    gemm_kernel<<<dim3(1024 / 128, 4096 / 128), blk, 0, stream>>>(
        kbf, wk2, 4096, 1024, 2, NK_, k2b, nullptr, nullptr, nullptr);

    diffattn_kernel<<<dim3(B_ * H_ * (NQ_ / 128)), blk, 0, stream>>>(
        q1b, q2b, k1b, k2b, v1t, lam1, lam2, xbuf);

    rmsnorm_kernel<<<dim3(2048), blk, 0, stream>>>(xbuf, norm_w, xn);

    gemm_kernel<<<dim3(1024 / 128, 2048 / 128), blk, 0, stream>>>(
        xn, wpr, 2048, 1024, 3, 1024, nullptr, nullptr, out, proj_b);
}
